// Agent_46248207843721
// MI455X (gfx1250) — compile-verified, matrix operand order checked by reading, not run on hardware
//
#include <hip/hip_runtime.h>
#include <hip/hip_bf16.h>
#include <math.h>

// ---------------- problem constants ----------------
#define NB    64          // segments
#define SEGL  2048        // segment length
#define NROW  (NB*SEGL)   // 131072 rows
#define DIN   128
#define HID   1024
#define KCAT  256         // concat(x_seeds, x_nodes) K
#define MT    64          // rows per workgroup tile
#define NWG   (NROW/MT)   // 2048 workgroups

typedef __attribute__((ext_vector_type(16))) __bf16 v16bf;
typedef __attribute__((ext_vector_type(8)))  __bf16 v8bf;
typedef __attribute__((ext_vector_type(8)))  float  v8f;

__device__ __forceinline__ unsigned short f2bfu(float f) {
  union { float f; unsigned u; } v; v.f = f;
  unsigned r = v.u + 0x7FFFu + ((v.u >> 16) & 1u);   // round-to-nearest-even
  return (unsigned short)(r >> 16);
}
__device__ __forceinline__ __bf16 f2bf(float f) {
  unsigned short h = f2bfu(f);
  __bf16 o; __builtin_memcpy(&o, &h, 2);
  return o;
}

// Load one 16x32 bf16 operand tile (A or B) in the CDNA5 16-bit WMMA layout:
// lane (l&15) = matrix row; half-wave (l>>4) = K-offset 8.
// Per lane: 8 bf16 at p0[0..7] (VGPR0-3), 8 bf16 at p0[16..23] (VGPR4-7).
// Union aliasing -> the two b128 loads land directly in the 8-VGPR operand.
__device__ __forceinline__ v16bf load_ab_tile(const __bf16* p0) {
  union { v8bf h[2]; v16bf v; } u;
  u.h[0] = *(const v8bf*)(p0);
  u.h[1] = *(const v8bf*)(p0 + 16);
  return u.v;
}

// One fused GEMM layer for a 64xK LDS tile: dst = act(src @ W^T + bias)
// Wave w handles rows [16*(w&3), +16), cols [512*(w>>2), +512) as 4 groups of
// eight 16x16 tiles; the A tile is reused across all 8 B tiles of a k-step.
__device__ __forceinline__ void layer_gemm(const __bf16* __restrict__ srcLds, int K,
                                           const __bf16* __restrict__ W,
                                           const float* __restrict__ bias,
                                           __bf16* __restrict__ dstLds, bool do_swish,
                                           int wave, int lane) {
  const int rt = wave & 3;
  const int colhalf = wave >> 2;
  const int m = lane & 15, half = lane >> 4;
  const __bf16* aBase = srcLds + (size_t)(rt * 16 + m) * K + half * 8;

  for (int cg = 0; cg < 4; ++cg) {
    const int nb0 = colhalf * 512 + cg * 128;
    v8f acc[8] = {};
    for (int ks = 0; ks < K; ks += 32) {
      v16bf a = load_ab_tile(aBase + ks);
#pragma unroll
      for (int t = 0; t < 8; ++t) {
        const __bf16* bp = W + (size_t)(nb0 + t * 16 + m) * K + ks + half * 8;
        v16bf b = load_ab_tile(bp);
        acc[t] = __builtin_amdgcn_wmma_f32_16x16x32_bf16(
            false, a, false, b, (short)0, acc[t], false, false);
      }
    }
#pragma unroll
    for (int t = 0; t < 8; ++t) {
      const int n = nb0 + t * 16 + m;
      const float bv = bias ? bias[n] : 0.0f;
#pragma unroll
      for (int r = 0; r < 8; ++r) {
        float v = acc[t][r] + bv;
        if (do_swish) v = v / (1.0f + __expf(-v));
        // C/D layout: VGPR r holds row (rt*16 + half*8 + r), col n
        dstLds[(size_t)(rt * 16 + half * 8 + r) * HID + n] = f2bf(v);
      }
    }
  }
}

// Final layer: never materialize h2; fold node-score dot and segment pooling
// into the WMMA epilogue via LDS atomics.
__device__ __forceinline__ void layer2_gemm(const __bf16* __restrict__ srcLds,
                                            const __bf16* __restrict__ W,
                                            const float* __restrict__ bias,
                                            const float* __restrict__ w_ns,
                                            float* __restrict__ lds_pool,
                                            float* __restrict__ lds_ns,
                                            int wave, int lane) {
  const int rt = wave & 3;
  const int colhalf = wave >> 2;
  const int m = lane & 15, half = lane >> 4;
  const __bf16* aBase = srcLds + (size_t)(rt * 16 + m) * HID + half * 8;

  for (int cg = 0; cg < 4; ++cg) {
    const int nb0 = colhalf * 512 + cg * 128;
    v8f acc[8] = {};
    for (int ks = 0; ks < HID; ks += 32) {
      v16bf a = load_ab_tile(aBase + ks);
#pragma unroll
      for (int t = 0; t < 8; ++t) {
        const __bf16* bp = W + (size_t)(nb0 + t * 16 + m) * HID + ks + half * 8;
        v16bf b = load_ab_tile(bp);
        acc[t] = __builtin_amdgcn_wmma_f32_16x16x32_bf16(
            false, a, false, b, (short)0, acc[t], false, false);
      }
    }
#pragma unroll
    for (int t = 0; t < 8; ++t) {
      const int n = nb0 + t * 16 + m;
      const float bv = bias[n];
      const float wns = w_ns[n];
      float psum = 0.0f;
#pragma unroll
      for (int r = 0; r < 8; ++r) {
        float v = acc[t][r] + bv;
        v = v / (1.0f + __expf(-v));           // swish
        psum += v;                             // column-sum for mean pool
        atomicAdd(&lds_ns[rt * 16 + half * 8 + r], v * wns);  // row dot w_ns
      }
      atomicAdd(&lds_pool[n], psum);
    }
  }
}

// ---------------- kernel 1a: weight prep (fp32 -> bf16, zero pool sums) ----
__global__ __launch_bounds__(256) void prep_w_kernel(
    const float* __restrict__ Ws, const float* __restrict__ Wn,
    const float* __restrict__ W1, const float* __restrict__ W2,
    __bf16* __restrict__ Wcat, __bf16* __restrict__ W1b, __bf16* __restrict__ W2b,
    float* __restrict__ stop_sums) {
  const int E0 = HID * KCAT;                 // 262144
  const int E1 = E0 + HID * HID;             // + 1M
  const int E2 = E1 + HID * HID;             // + 1M
  const int E3 = E2 + NB * HID;              // + 64K zeros
  int i = blockIdx.x * 256 + threadIdx.x;
  if (i < E0) {
    int n = i >> 8, d = i & 255;
    float v = (d < DIN) ? Ws[n * DIN + d] : Wn[n * DIN + (d - DIN)];
    Wcat[i] = f2bf(v);
  } else if (i < E1) {
    W1b[i - E0] = f2bf(W1[i - E0]);
  } else if (i < E2) {
    W2b[i - E1] = f2bf(W2[i - E1]);
  } else if (i < E3) {
    stop_sums[i - E2] = 0.0f;
  }
}

// ---------------- kernel 1b: x prep -> bf16 concat [N, 256] ---------------
__global__ __launch_bounds__(256) void prep_x_kernel(
    const float* __restrict__ xs, const float* __restrict__ xn,
    __bf16* __restrict__ xcat) {
  const int total = NROW * 64;               // float4 units
  uint2* dst = (uint2*)xcat;
  const float4* xs4 = (const float4*)xs;
  const float4* xn4 = (const float4*)xn;
  for (int u = blockIdx.x * 256 + threadIdx.x; u < total; u += gridDim.x * 256) {
    int row = u >> 6, c = u & 63;
    float4 f = (c < 32) ? xs4[(size_t)row * 32 + c]
                        : xn4[(size_t)row * 32 + (c - 32)];
    uint2 p;
    p.x = (unsigned)f2bfu(f.x) | ((unsigned)f2bfu(f.y) << 16);
    p.y = (unsigned)f2bfu(f.z) | ((unsigned)f2bfu(f.w) << 16);
    dst[u] = p;                              // one 8B store per float4
  }
}

// ---------------- kernel 2: fused 3-layer MLP + score + pool ---------------
__global__ __launch_bounds__(256) void fused_mlp_kernel(
    const __bf16* __restrict__ xcat,
    const __bf16* __restrict__ Wcat, const __bf16* __restrict__ W1b,
    const __bf16* __restrict__ W2b,
    const float* __restrict__ b1, const float* __restrict__ b2,
    const float* __restrict__ w_ns,
    float* __restrict__ node_scores, float* __restrict__ stop_sums) {
  extern __shared__ __align__(16) char smem[];
  __bf16* lds_x  = (__bf16*)smem;                 // 64 x 256   (32 KB)
  __bf16* lds_h0 = lds_x + MT * KCAT;             // 64 x 1024  (128 KB)
  __bf16* lds_h1 = lds_h0 + MT * HID;             // 64 x 1024  (128 KB)
  float*  lds_pool = (float*)(lds_h1 + MT * HID); // 1024
  float*  lds_ns   = lds_pool + HID;              // 64

  const int tid  = threadIdx.x;
  const int lane = tid & 31;
  const int wave = tid >> 5;
  const int row0 = blockIdx.x * MT;

  // ---- async-stage the contiguous 32KB x tile into LDS (ASYNCcnt path) ----
  {
    const char* gsrc = (const char*)(xcat + (size_t)row0 * KCAT);
    unsigned ldsOff = (unsigned)(size_t)(void*)lds_x;   // low 32 bits = LDS addr
#pragma unroll
    for (int k = 0; k < 8; ++k) {
      int i = tid + k * 256;                            // 2048 x 16B chunks
      unsigned l = ldsOff + (unsigned)i * 16u;
      const char* g = gsrc + (size_t)i * 16;
      asm volatile("global_load_async_to_lds_b128 %0, %1, off"
                   :: "v"(l), "v"(g) : "memory");
    }
  }

  // warm L2 for the weight stream (global_prefetch_b8)
  __builtin_prefetch(W1b + (size_t)tid * 8192, 0, 1);
  __builtin_prefetch(W2b + (size_t)tid * 8192, 0, 1);

  for (int i = tid; i < HID; i += 256) lds_pool[i] = 0.0f;
  if (tid < MT) lds_ns[tid] = 0.0f;

  asm volatile("s_wait_asynccnt 0x0" ::: "memory");
  __syncthreads();

  layer_gemm(lds_x,  KCAT, Wcat, nullptr, lds_h0, false, wave, lane);  // embedding
  __syncthreads();
  layer_gemm(lds_h0, HID,  W1b,  b1,      lds_h1, true,  wave, lane);  // swish layer 1
  __syncthreads();
  layer2_gemm(lds_h1, W2b, b2, w_ns, lds_pool, lds_ns, wave, lane);    // swish layer 2 + epilogue
  __syncthreads();

  if (tid < MT) node_scores[row0 + tid] = lds_ns[tid];
  const int seg = row0 / SEGL;
  for (int i = tid; i < HID; i += 256)
    atomicAdd(&stop_sums[(size_t)seg * HID + i], lds_pool[i]);
}

// ---------------- kernel 3: per-segment softmaxes + output -----------------
__global__ __launch_bounds__(256) void finalize_kernel(
    const float* __restrict__ node_scores, const float* __restrict__ stop_sums,
    const float* __restrict__ W_stop, const int* __restrict__ indptr,
    float* __restrict__ out) {
  __shared__ float r0[256], r1[256];
  const int b = blockIdx.x, tid = threadIdx.x;
  const int start = indptr[b * 3];

  // stop logits from mean pool
  float d0 = 0.0f, d1 = 0.0f;
  for (int n = tid; n < HID; n += 256) {
    float m = stop_sums[(size_t)b * HID + n] * (1.0f / (float)SEGL);
    d0 += m * W_stop[n];
    d1 += m * W_stop[HID + n];
  }
  r0[tid] = d0; r1[tid] = d1; __syncthreads();
  for (int s = 128; s > 0; s >>= 1) {
    if (tid < s) { r0[tid] += r0[tid + s]; r1[tid] += r1[tid + s]; }
    __syncthreads();
  }
  const float l0 = r0[0], l1 = r1[0];
  const float m2 = fmaxf(l0, l1);
  const float lse2 = m2 + logf(__expf(l0 - m2) + __expf(l1 - m2));
  const float stop0 = l0 - lse2, stop1 = l1 - lse2;
  __syncthreads();

  // log-softmax over the segment's node scores
  float mx = -3.402823466e38f;
  for (int i = tid; i < SEGL; i += 256) mx = fmaxf(mx, node_scores[start + i]);
  r0[tid] = mx; __syncthreads();
  for (int s = 128; s > 0; s >>= 1) {
    if (tid < s) r0[tid] = fmaxf(r0[tid], r0[tid + s]);
    __syncthreads();
  }
  const float gmax = r0[0];
  __syncthreads();
  float se = 0.0f;
  for (int i = tid; i < SEGL; i += 256) se += __expf(node_scores[start + i] - gmax);
  r1[tid] = se; __syncthreads();
  for (int s = 128; s > 0; s >>= 1) {
    if (tid < s) r1[tid] += r1[tid + s];
    __syncthreads();
  }
  const float lse = gmax + logf(r1[0]);

  float* orow = out + (size_t)b * (SEGL + 1);
  for (int i = tid; i < SEGL; i += 256)
    orow[i] = node_scores[start + i] - lse + stop0;
  if (tid == 0) orow[SEGL] = stop1;
}

// ---------------- host launcher --------------------------------------------
extern "C" void kernel_launch(void* const* d_in, const int* in_sizes, int n_in,
                              void* d_out, int out_size, void* d_ws, size_t ws_size,
                              hipStream_t stream) {
  const float* xs     = (const float*)d_in[0];
  const float* xn     = (const float*)d_in[1];
  const float* W_seed = (const float*)d_in[2];
  const float* W_node = (const float*)d_in[3];
  const float* W1     = (const float*)d_in[4];
  const float* b1     = (const float*)d_in[5];
  const float* W2     = (const float*)d_in[6];
  const float* b2     = (const float*)d_in[7];
  const float* w_ns   = (const float*)d_in[8];
  const float* W_stop = (const float*)d_in[9];
  const int*   indptr = (const int*)d_in[10];
  float* out = (float*)d_out;

  // workspace carve-out (all 256B-aligned)
  char* ws = (char*)d_ws;
  size_t off = 0;
  __bf16* Wcat = (__bf16*)(ws + off); off += (size_t)HID * KCAT * 2;
  __bf16* W1b  = (__bf16*)(ws + off); off += (size_t)HID * HID * 2;
  __bf16* W2b  = (__bf16*)(ws + off); off += (size_t)HID * HID * 2;
  float* node_scores = (float*)(ws + off); off += (size_t)NROW * 4;
  float* stop_sums   = (float*)(ws + off); off += (size_t)NB * HID * 4;
  __bf16* xcat = (__bf16*)(ws + off);  off += (size_t)NROW * KCAT * 2;  // 67MB
  (void)ws_size; (void)in_sizes; (void)n_in; (void)out_size;

  const int prepElems = HID * KCAT + 2 * HID * HID + NB * HID;
  prep_w_kernel<<<(prepElems + 255) / 256, 256, 0, stream>>>(
      W_seed, W_node, W1, W2, Wcat, W1b, W2b, stop_sums);
  prep_x_kernel<<<8192, 256, 0, stream>>>(xs, xn, xcat);

  const size_t smemBytes =
      (size_t)(MT * KCAT + 2 * MT * HID) * sizeof(__bf16) +
      (size_t)(HID + MT) * sizeof(float);   // 299,264 B of the 320KB WGP LDS
  (void)hipFuncSetAttribute((const void*)fused_mlp_kernel,
                            hipFuncAttributeMaxDynamicSharedMemorySize,
                            (int)smemBytes);
  fused_mlp_kernel<<<NWG, 256, smemBytes, stream>>>(
      xcat, Wcat, W1b, W2b, b1, b2, w_ns, node_scores, stop_sums);

  finalize_kernel<<<NB, 256, 0, stream>>>(
      node_scores, stop_sums, W_stop, indptr, out);
}
